// sparseTransformer_22866405884072
// MI455X (gfx1250) — matrix operand order repaired and checked by reading, not compiled, not run on hardware
//
#include <hip/hip_runtime.h>
#include <hip/hip_bf16.h>

// ---------------------------------------------------------------------------
// CDNA5 / gfx1250 sparse-transformer forward. bf16 data path, f32 accumulate,
// matmuls via v_wmma_f32_16x16x32_bf16, GEMM tiles fed by the Tensor Data
// Mover (tensor_load_to_lds / TENSORcnt), attention K staged via async-LDS.
// ---------------------------------------------------------------------------

typedef __bf16 bf16;
typedef __attribute__((ext_vector_type(16))) __bf16 v16bf;
typedef __attribute__((ext_vector_type(8)))  float  v8f;
typedef __attribute__((ext_vector_type(4)))  unsigned u32x4;
typedef __attribute__((ext_vector_type(8)))  int      i32x8;
typedef __attribute__((ext_vector_type(4)))  int      i32x4;

constexpr int N_PTS  = 16384;
constexpr int DMODEL = 512;
constexpr int BATCH  = 8;
constexpr int HEADS  = 8;
constexpr int HD     = 64;
constexpr int LAYERS = 6;
constexpr int QKV_LD = 3 * DMODEL;

union FragB { v16bf v; uint4 u[2]; };

__device__ __forceinline__ v8f wmma_bf16(v16bf a, v16bf b, v8f c) {
  // 8 args: (neg_a, A, neg_b, B, c_mod, C, reuse_a, reuse_b)
  return __builtin_amdgcn_wmma_f32_16x16x32_bf16(false, a, false, b, (short)0, c,
                                                 false, false);
}

// Async global -> LDS copy (16B), tracked by ASYNCcnt (ISA 08_async_tensor §4)
__device__ __forceinline__ void async_ld_b128(unsigned ldsOff, const void* gp) {
  asm volatile("global_load_async_to_lds_b128 %0, %1, off"
               :: "v"(ldsOff), "v"(gp) : "memory");
}
__device__ __forceinline__ void wait_async0() {
  asm volatile("s_wait_asynccnt 0x0" ::: "memory");
}

// ---------------------------------------------------------------------------
// Tensor Data Mover: 2D tile (tileX elems of 2B per row, tileY rows) from a
// row-major tensor with row stride strideX elems into LDS at ldsOff, padding
// the LDS destination by padAmount DWORDs every padInterval DWORDs.
// D# layout per ISA 08_async_tensor §8.3/8.4 (group0: count/lds/global/type,
// group1: data_size, pad, dims, dim0 stride). Groups 2/3 zero (2D tensor).
// ---------------------------------------------------------------------------
__device__ __forceinline__ void tdm_load_2d(
    unsigned ldsOff, const void* gaddr,
    unsigned tileX, unsigned tileY,
    unsigned tensorX, unsigned tensorY, unsigned strideX,
    unsigned padIntervalCode, unsigned padAmountCode)
{
  const unsigned long long ga = (unsigned long long)(uintptr_t)gaddr;
  u32x4 g0;
  g0[0] = 1u;                                          // count=1 (valid), user
  g0[1] = ldsOff;                                      // lds_addr (bytes)
  g0[2] = (unsigned)(ga & 0xFFFFFFFFu);                // global_addr[31:0]
  g0[3] = (unsigned)((ga >> 32) & 0x01FFFFFFu)         // global_addr[56:32]
        | (2u << 30);                                  // type = 2 ("image")
  const unsigned w0 = (1u << 16)                       // data_size = 1 -> 2B
                    | (1u << 20)                       // pad_enable
                    | (padIntervalCode << 22)          // pad_interval
                    | (padAmountCode << 25);           // pad_amount
  i32x8 g1;
  g1[0] = (int)w0;
  g1[1] = (int)((tensorX & 0xFFFFu) << 16);            // tensor_dim0 @ [79:48]
  g1[2] = (int)((tensorX >> 16) | ((tensorY & 0xFFFFu) << 16)); // dim1 @[111:80]
  g1[3] = (int)((tensorY >> 16) | (tileX << 16));      // tile_dim0 @ [127:112]
  g1[4] = (int)(tileY & 0xFFFFu);                      // tile_dim1; tile_dim2=0
  g1[5] = (int)strideX;                                // dim0_stride[31:0]
  g1[6] = 0;                                           // stride hi / dim1_stride
  g1[7] = 0;
  const i32x4 z4 = {0, 0, 0, 0};
#if defined(__clang_major__) && __clang_major__ >= 23
  const i32x8 z8 = {0, 0, 0, 0, 0, 0, 0, 0};
  __builtin_amdgcn_tensor_load_to_lds(g0, g1, z4, z4, z8, 0);
#else
  __builtin_amdgcn_tensor_load_to_lds(g0, g1, z4, z4, 0);
#endif
}

// ---------------------------------------------------------------------------
// Segment offsets from sorted batch_idx (lower-bound binary search per batch)
// ---------------------------------------------------------------------------
__global__ void offsets_kernel(const int* __restrict__ bidx, int* __restrict__ offs) {
  const int b = threadIdx.x;
  if (b > BATCH) return;
  int lo = 0, hi = N_PTS;
  while (lo < hi) {
    const int mid = (lo + hi) >> 1;
    if (bidx[mid] < b) lo = mid + 1; else hi = mid;
  }
  offs[b] = lo;
}

// ---------------------------------------------------------------------------
// f32 -> bf16 conversion (elementwise / transposed DxD for w_in, w_out)
// ---------------------------------------------------------------------------
__global__ void cvt_kernel(const float* __restrict__ in, bf16* __restrict__ out,
                           size_t n) {
  size_t i = (size_t)blockIdx.x * blockDim.x + threadIdx.x;
  const size_t stride = (size_t)gridDim.x * blockDim.x;
  for (; i < n; i += stride) out[i] = (bf16)in[i];
}

__global__ void cvtT_kernel(const float* __restrict__ in, bf16* __restrict__ out) {
  const int idx = blockIdx.x * 256 + threadIdx.x;  // DMODEL*DMODEL elements
  const int n = idx >> 9;
  const int k = idx & (DMODEL - 1);
  out[idx] = (bf16)in[(size_t)k * DMODEL + n];     // out[n][k] = in[k][n]
}

__global__ void add_kernel(const bf16* __restrict__ a, const bf16* __restrict__ b,
                           bf16* __restrict__ c, size_t n) {
  size_t i = (size_t)blockIdx.x * blockDim.x + threadIdx.x;
  const size_t stride = (size_t)gridDim.x * blockDim.x;
  for (; i < n; i += stride) c[i] = (bf16)((float)a[i] + (float)b[i]);
}

// ---------------------------------------------------------------------------
// GEMM: C[M x Nout] = A[M x K](bf16) * Bw[Nout x K](bf16 row-major)^T + bias
// 128x128 tile / 8 waves (each 32x64 = 2x4 WMMA tiles), K step 32.
// Double-buffered LDS filled by the TDM: rows of 16 DWORDs padded by 8 DWORDs
// (pad_interval code 3 = 16 DW, pad_amount code 7 = 8 DW) -> LDT = 48 elems.
// ---------------------------------------------------------------------------
constexpr int BLK   = 128;
constexpr int BLK_K = 32;
constexpr int LDT   = 48;   // 96B rows -> 16B aligned, matches TDM padding

template <bool SILU, bool OUTF32>
__global__ __launch_bounds__(256) void gemm_kernel(
    const bf16* __restrict__ A, const bf16* __restrict__ Bw,
    const float* __restrict__ bias, void* __restrict__ Cout,
    int M, int K, int Nout)
{
  __shared__ bf16 sA[2][BLK * LDT];
  __shared__ bf16 sB[2][BLK * LDT];
  const int tid   = threadIdx.x;
  const int lane  = tid & 31;
  const int wave  = tid >> 5;
  const int waveM = wave >> 1;   // 0..3
  const int waveN = wave & 1;    // 0..1
  const int mBase = blockIdx.x * BLK;
  const int nBase = blockIdx.y * BLK;
  const int m  = lane & 15;
  const int kh = lane >> 4;

  v8f acc[2][4];
#pragma unroll
  for (int i = 0; i < 2; ++i)
#pragma unroll
    for (int j = 0; j < 4; ++j)
#pragma unroll
      for (int r = 0; r < 8; ++r) acc[i][j][r] = 0.f;

  const unsigned ldsA[2] = {(unsigned)(uintptr_t)&sA[0][0],
                            (unsigned)(uintptr_t)&sA[1][0]};
  const unsigned ldsB[2] = {(unsigned)(uintptr_t)&sB[0][0],
                            (unsigned)(uintptr_t)&sB[1][0]};
  const bf16* Atile = A  + (size_t)mBase * K;
  const bf16* Btile = Bw + (size_t)nBase * K;

  // prologue: TDM DMA of tile 0 into buffer 0 (wave 0 issues; EXEC-independent)
  if (wave == 0) {
    tdm_load_2d(ldsA[0], Atile, BLK_K, BLK, K, M,    K, 3, 7);
    tdm_load_2d(ldsB[0], Btile, BLK_K, BLK, K, Nout, K, 3, 7);
  }

  const int nT = K / BLK_K;
  for (int i = 0; i < nT; ++i) {
    const int cur = i & 1;
    if (wave == 0) {
      if (i + 1 < nT) {   // DMA next tile while this one is consumed
        const int k1 = (i + 1) * BLK_K;
        tdm_load_2d(ldsA[cur ^ 1], Atile + k1, BLK_K, BLK, K, M,    K, 3, 7);
        tdm_load_2d(ldsB[cur ^ 1], Btile + k1, BLK_K, BLK, K, Nout, K, 3, 7);
        __builtin_amdgcn_s_wait_tensorcnt(2);   // tile i complete (in-order)
      } else {
        __builtin_amdgcn_s_wait_tensorcnt(0);
      }
    }
    __syncthreads();
    if (i + 2 < nT && tid < BLK) {              // global_prefetch_b8 -> L2
      __builtin_prefetch(Atile + (size_t)tid * K + (i + 2) * BLK_K, 0, 1);
      __builtin_prefetch(Btile + (size_t)tid * K + (i + 2) * BLK_K, 0, 1);
    }
    FragB af[2];
#pragma unroll
    for (int t = 0; t < 2; ++t) {
      const bf16* p = &sA[cur][(waveM * 32 + t * 16 + m) * LDT + kh * 8];
      af[t].u[0] = *reinterpret_cast<const uint4*>(p);
      af[t].u[1] = *reinterpret_cast<const uint4*>(p + 16);
    }
    FragB bfr[4];
#pragma unroll
    for (int t = 0; t < 4; ++t) {
      const bf16* p = &sB[cur][(waveN * 64 + t * 16 + m) * LDT + kh * 8];
      bfr[t].u[0] = *reinterpret_cast<const uint4*>(p);
      bfr[t].u[1] = *reinterpret_cast<const uint4*>(p + 16);
    }
#pragma unroll
    for (int tm = 0; tm < 2; ++tm)
#pragma unroll
      for (int tn = 0; tn < 4; ++tn)
        acc[tm][tn] = wmma_bf16(af[tm].v, bfr[tn].v, acc[tm][tn]);
    if (i + 1 < nT) __syncthreads();  // done reading before TDM overwrites nxt
  }

  // epilogue: C layout row = r + 8*kh, col = lane&15
#pragma unroll
  for (int tn = 0; tn < 4; ++tn) {
    const int col = nBase + waveN * 64 + tn * 16 + m;
    const float bv = bias ? bias[col] : 0.f;
#pragma unroll
    for (int tm = 0; tm < 2; ++tm) {
      const int rowB = mBase + waveM * 32 + tm * 16 + kh * 8;
#pragma unroll
      for (int r = 0; r < 8; ++r) {
        float v = acc[tm][tn][r] + bv;
        if (SILU) v = v / (1.f + __expf(-v));
        const size_t idx = (size_t)(rowB + r) * Nout + col;
        if (OUTF32) reinterpret_cast<float*>(Cout)[idx] = v;
        else        reinterpret_cast<bf16*>(Cout)[idx]  = (bf16)v;
      }
    }
  }
}

// ---------------------------------------------------------------------------
// Flash attention over ragged segments. 4 waves / block; each wave owns a
// 16-query tile, all share the staged 32-key K/V tiles (K via async-LDS DMA).
// ---------------------------------------------------------------------------
constexpr int KLD  = 72;  // sK row stride (elems): 144B -> 16B aligned
constexpr int VTLD = 40;  // sVt row stride (elems): 80B -> 16B aligned

__global__ __launch_bounds__(128) void attn_kernel(
    const bf16* __restrict__ qkv, bf16* __restrict__ outO,
    const int* __restrict__ offs)
{
  __shared__ bf16 sK[32 * KLD];
  __shared__ bf16 sVt[64 * VTLD];
  __shared__ bf16 sP[4][16 * 32];
  const int b  = blockIdx.z;
  const int hh = blockIdx.y;
  const int s0 = offs[b];
  const int L  = offs[b + 1] - s0;
  const int blockBase = blockIdx.x * 64;
  if (L <= 0 || blockBase >= L) return;   // uniform exit

  const int tid  = threadIdx.x;
  const int lane = tid & 31;
  const int wave = tid >> 5;
  const int m  = lane & 15;
  const int kh = lane >> 4;
  const int qBase = blockBase + wave * 16;   // may exceed L; compute clamped,
                                             // stores guarded (EXEC stays full)
  // Q as A-fragments over two K=32 chunks of head-dim 64
  FragB aQ[2];
  {
    const int q = qBase + m;
    const int qRow = s0 + (q < L ? q : L - 1);
    const bf16* qp = qkv + (size_t)qRow * QKV_LD + hh * HD;
#pragma unroll
    for (int kk = 0; kk < 2; ++kk) {
      const bf16* p = qp + kk * 32 + kh * 8;
      aQ[kk].u[0] = *reinterpret_cast<const uint4*>(p);
      aQ[kk].u[1] = *reinterpret_cast<const uint4*>(p + 16);
    }
  }

  v8f acc[4];
#pragma unroll
  for (int t = 0; t < 4; ++t)
#pragma unroll
    for (int r = 0; r < 8; ++r) acc[t][r] = 0.f;
  float rowM[8], rowL[8];
#pragma unroll
  for (int r = 0; r < 8; ++r) { rowM[r] = -1e30f; rowL[r] = 0.f; }

  const int vKey = tid >> 2;   // V staging: key row handled by this thread
  const int dq   = tid & 3;    // 16-dim quarter of that key row

  const int nKB = (L + 31) >> 5;
  for (int jb = 0; jb < nKB; ++jb) {
    const int jBase = jb * 32;
    // ---- stage K tile via async DMA: 32 rows x 128B = 256 16B chunks ----
#pragma unroll
    for (int j = 0; j < 2; ++j) {
      const int c = tid + j * 128;
      const int row = c >> 3, ch = c & 7;
      const int kj = jBase + row;
      const int kc = kj < L ? kj : L - 1;
      const bf16* gp = qkv + (size_t)(s0 + kc) * QKV_LD + DMODEL + hh * HD + ch * 8;
      async_ld_b128((unsigned)(uintptr_t)&sK[row * KLD + ch * 8], gp);
    }
    // ---- stage V tile transposed (VALU; masked keys have p==0) ----
    {
      const int kj = jBase + vKey;
      const int vc = kj < L ? kj : L - 1;
      const bf16* vp = qkv + (size_t)(s0 + vc) * QKV_LD + 2 * DMODEL + hh * HD + dq * 16;
      union { uint4 u[2]; bf16 e[16]; } vr;
      vr.u[0] = reinterpret_cast<const uint4*>(vp)[0];
      vr.u[1] = reinterpret_cast<const uint4*>(vp)[1];
#pragma unroll
      for (int i = 0; i < 16; ++i)
        sVt[(dq * 16 + i) * VTLD + vKey] = vr.e[i];
    }
    wait_async0();
    __syncthreads();

    // ---- S = Q K^T for the two 16-key sub-tiles ----
    v8f st[2];
#pragma unroll
    for (int kb = 0; kb < 2; ++kb) {
      FragB bk[2];
#pragma unroll
      for (int kk = 0; kk < 2; ++kk) {
        const bf16* p = &sK[(kb * 16 + m) * KLD + kk * 32 + kh * 8];
        bk[kk].u[0] = *reinterpret_cast<const uint4*>(p);
        bk[kk].u[1] = *reinterpret_cast<const uint4*>(p + 16);
      }
      v8f s;
#pragma unroll
      for (int r = 0; r < 8; ++r) s[r] = 0.f;
      s = wmma_bf16(aQ[0].v, bk[0].v, s);
      s = wmma_bf16(aQ[1].v, bk[1].v, s);
      const bool valid = (jBase + kb * 16 + m) < L;   // key-padding mask
#pragma unroll
      for (int r = 0; r < 8; ++r) s[r] = valid ? s[r] * 0.125f : -1e30f;
      st[kb] = s;
    }
    // ---- online softmax (rows r+8*kh across VGPRs, cols across 16 lanes) ----
    bf16* myP = sP[wave];
#pragma unroll
    for (int r = 0; r < 8; ++r) {
      float v0 = st[0][r], v1 = st[1][r];
      float ml = fmaxf(v0, v1);
#pragma unroll
      for (int off = 1; off < 16; off <<= 1) ml = fmaxf(ml, __shfl_xor(ml, off));
      const float mNew = fmaxf(rowM[r], ml);
      const float p0 = __expf(v0 - mNew);
      const float p1 = __expf(v1 - mNew);
      float ls = p0 + p1;
#pragma unroll
      for (int off = 1; off < 16; off <<= 1) ls += __shfl_xor(ls, off);
      const float corr = __expf(rowM[r] - mNew);
      rowL[r] = rowL[r] * corr + ls;
      rowM[r] = mNew;
#pragma unroll
      for (int tn = 0; tn < 4; ++tn) acc[tn][r] *= corr;
      const int prow = r + kh * 8;
      myP[prow * 32 + m]      = (bf16)p0;
      myP[prow * 32 + 16 + m] = (bf16)p1;
    }
    // ---- O += P V (per-wave P, shared V^T; same-wave LDS is in-order) ----
    FragB aP;
    {
      const bf16* p = &myP[m * 32 + kh * 8];
      aP.u[0] = *reinterpret_cast<const uint4*>(p);
      aP.u[1] = *reinterpret_cast<const uint4*>(p + 16);
    }
#pragma unroll
    for (int tn = 0; tn < 4; ++tn) {
      const bf16* p = &sVt[(tn * 16 + m) * VTLD + kh * 8];
      FragB bV;
      bV.u[0] = *reinterpret_cast<const uint4*>(p);
      bV.u[1] = *reinterpret_cast<const uint4*>(p + 16);
      acc[tn] = wmma_bf16(aP.v, bV.v, acc[tn]);
    }
    __syncthreads();   // all waves done with sK/sVt before restaging
  }

#pragma unroll
  for (int r = 0; r < 8; ++r) {
    const int qr = qBase + r + kh * 8;
    if (qr >= L) continue;
    const float inv = 1.f / rowL[r];
#pragma unroll
    for (int tn = 0; tn < 4; ++tn)
      outO[(size_t)(s0 + qr) * DMODEL + hh * HD + tn * 16 + m] =
          (bf16)(acc[tn][r] * inv);
  }
}

// ---------------------------------------------------------------------------
// out = LayerNorm(x + y) * g + b ; one wave per row, f32 math, bf16 out
// ---------------------------------------------------------------------------
__global__ __launch_bounds__(256) void ln_kernel(
    const bf16* __restrict__ x, const bf16* __restrict__ y,
    const float* __restrict__ g, const float* __restrict__ bta,
    bf16* __restrict__ out)
{
  const int wave = threadIdx.x >> 5;
  const int lane = threadIdx.x & 31;
  const size_t row = (size_t)blockIdx.x * 8 + wave;
  const bf16* xr = x + row * DMODEL;
  const bf16* yr = y + row * DMODEL;
  float v[16], s = 0.f, s2 = 0.f;
#pragma unroll
  for (int i = 0; i < 16; ++i) {
    const int c = lane + i * 32;
    const float t = (float)xr[c] + (float)yr[c];
    v[i] = t; s += t; s2 += t * t;
  }
#pragma unroll
  for (int off = 1; off < 32; off <<= 1) {
    s  += __shfl_xor(s, off);
    s2 += __shfl_xor(s2, off);
  }
  const float mu  = s * (1.f / DMODEL);
  const float var = s2 * (1.f / DMODEL) - mu * mu;
  const float inv = rsqrtf(var + 1e-5f);
  bf16* orow = out + row * DMODEL;
#pragma unroll
  for (int i = 0; i < 16; ++i) {
    const int c = lane + i * 32;
    orow[c] = (bf16)((v[i] - mu) * inv * g[c] + bta[c]);
  }
}

// ---------------------------------------------------------------------------
extern "C" void kernel_launch(void* const* d_in, const int* in_sizes, int n_in,
                              void* d_out, int out_size, void* d_ws, size_t ws_size,
                              hipStream_t stream)
{
  (void)in_sizes; (void)n_in; (void)out_size; (void)ws_size;
  const float* features   = (const float*)d_in[0];
  const int*   batch_idx  = (const int*)  d_in[1];
  const float* w_in       = (const float*)d_in[2];
  const float* w_out      = (const float*)d_in[3];
  const float* in_proj_w  = (const float*)d_in[4];
  const float* in_proj_b  = (const float*)d_in[5];
  const float* out_proj_w = (const float*)d_in[6];
  const float* out_proj_b = (const float*)d_in[7];
  const float* ln1_g      = (const float*)d_in[8];
  const float* ln1_b      = (const float*)d_in[9];
  const float* lin1_w     = (const float*)d_in[10];
  const float* lin1_b     = (const float*)d_in[11];
  const float* lin2_w     = (const float*)d_in[12];
  const float* lin2_b     = (const float*)d_in[13];
  const float* ln2_g      = (const float*)d_in[14];
  const float* ln2_b      = (const float*)d_in[15];

  const size_t ND = (size_t)N_PTS * DMODEL;
  const size_t DD = (size_t)DMODEL * DMODEL;

  char* p = (char*)d_ws;
  auto carve = [&p](size_t bytes) -> char* {
    char* r = p;
    p += (bytes + 255) & ~(size_t)255;
    return r;
  };
  int*  offs   = (int*) carve((BATCH + 1) * sizeof(int));
  bf16* featb  = (bf16*)carve(ND * 2);
  bf16* xinb   = (bf16*)carve(ND * 2);
  bf16* xb     = (bf16*)carve(ND * 2);
  bf16* qkvb   = (bf16*)carve(ND * 3 * 2);
  bf16* attb   = (bf16*)carve(ND * 2);
  bf16* hb     = (bf16*)carve(ND * 2);
  bf16* tmpb   = (bf16*)carve(ND * 2);
  bf16* winTb  = (bf16*)carve(DD * 2);
  bf16* woutTb = (bf16*)carve(DD * 2);
  bf16* inwb   = (bf16*)carve((size_t)LAYERS * 3 * DD * 2);
  bf16* outwb  = (bf16*)carve((size_t)LAYERS * DD * 2);
  bf16* l1wb   = (bf16*)carve((size_t)LAYERS * DD * 2);
  bf16* l2wb   = (bf16*)carve((size_t)LAYERS * DD * 2);

  offsets_kernel<<<1, 32, 0, stream>>>(batch_idx, offs);
  cvt_kernel<<<4096, 256, 0, stream>>>(features,   featb, ND);
  cvt_kernel<<<4096, 256, 0, stream>>>(in_proj_w,  inwb,  (size_t)LAYERS * 3 * DD);
  cvt_kernel<<<4096, 256, 0, stream>>>(out_proj_w, outwb, (size_t)LAYERS * DD);
  cvt_kernel<<<4096, 256, 0, stream>>>(lin1_w,     l1wb,  (size_t)LAYERS * DD);
  cvt_kernel<<<4096, 256, 0, stream>>>(lin2_w,     l2wb,  (size_t)LAYERS * DD);
  cvtT_kernel<<<DD / 256, 256, 0, stream>>>(w_in,  winTb);
  cvtT_kernel<<<DD / 256, 256, 0, stream>>>(w_out, woutTb);

  const dim3 gThin(N_PTS / BLK, DMODEL / BLK);        // (128, 4)
  const dim3 gQkv (N_PTS / BLK, 3 * DMODEL / BLK);    // (128, 12)
  const dim3 gAttn(N_PTS / 64, HEADS, BATCH);         // 4 query tiles / block

  // x_in = features @ w_in
  gemm_kernel<false, false><<<gThin, 256, 0, stream>>>(
      featb, winTb, nullptr, xinb, N_PTS, DMODEL, DMODEL);
  hipMemcpyAsync(xb, xinb, ND * 2, hipMemcpyDeviceToDevice, stream);

  for (int l = 0; l < LAYERS; ++l) {
    gemm_kernel<false, false><<<gQkv, 256, 0, stream>>>(
        xb, inwb + (size_t)l * 3 * DD, in_proj_b + (size_t)l * 3 * DMODEL,
        qkvb, N_PTS, DMODEL, 3 * DMODEL);
    attn_kernel<<<gAttn, 128, 0, stream>>>(qkvb, attb, offs);
    gemm_kernel<false, false><<<gThin, 256, 0, stream>>>(
        attb, outwb + (size_t)l * DD, out_proj_b + (size_t)l * DMODEL,
        tmpb, N_PTS, DMODEL, DMODEL);
    ln_kernel<<<N_PTS / 8, 256, 0, stream>>>(
        xb, tmpb, ln1_g + (size_t)l * DMODEL, ln1_b + (size_t)l * DMODEL, xb);
    gemm_kernel<true, false><<<gThin, 256, 0, stream>>>(
        xb, l1wb + (size_t)l * DD, lin1_b + (size_t)l * DMODEL,
        hb, N_PTS, DMODEL, DMODEL);
    gemm_kernel<false, false><<<gThin, 256, 0, stream>>>(
        hb, l2wb + (size_t)l * DD, lin2_b + (size_t)l * DMODEL,
        tmpb, N_PTS, DMODEL, DMODEL);
    ln_kernel<<<N_PTS / 8, 256, 0, stream>>>(
        xb, tmpb, ln2_g + (size_t)l * DMODEL, ln2_b + (size_t)l * DMODEL, xb);
  }

  // out = (x + x_in) @ w_out   (f32 output)
  add_kernel<<<4096, 256, 0, stream>>>(xb, xinb, tmpb, ND);
  gemm_kernel<false, true><<<gThin, 256, 0, stream>>>(
      tmpb, woutTb, nullptr, d_out, N_PTS, DMODEL, DMODEL);
}
